// SubgraphSelectModule_24335284699144
// MI455X (gfx1250) — matrix operand order repaired and checked by reading, not compiled, not run on hardware
//
#include <hip/hip_runtime.h>
#include <hip/hip_bf16.h>
#include <math.h>

// ---------------- problem constants (from reference) ----------------
#define G_      64
#define S_      64
#define NPER_   32
#define KTOP_   52                      // ceil(0.8*64)
#define NFEAT_  512
#define HID_    1024
#define NCLASS_ 256
#define NSUBS_  (G_ * S_)               // 4096
#define NNODES_ (NSUBS_ * NPER_)        // 131072
#define E_      2097152
#define NEWSUBS_ (G_ * KTOP_)           // 3328
#define NEWN_    (NEWSUBS_ * NPER_)     // 106496
#define MROWS_   32                     // rows per workgroup in fused MLP

// output float offsets (concatenated flat, return order)
#define O_NEWX  0UL
#define O_ZNODE ((size_t)NEWN_ * NFEAT_)
#define O_ZSUB  (O_ZNODE + (size_t)NEWN_ * NCLASS_)
#define O_EDGE  (O_ZSUB + (size_t)NEWSUBS_ * NCLASS_)
#define O_BATCH (O_EDGE + 2UL * E_)
#define O_BELON (O_BATCH + (size_t)NEWN_)
#define O_MAP   (O_BELON + (size_t)NEWSUBS_)
#define O_SCORE (O_MAP + (size_t)NEWN_)

// async global->LDS staging if the toolchain exposes the gfx1250 builtin
#if defined(__has_builtin)
#if __has_builtin(__builtin_amdgcn_global_load_async_to_lds_b128)
#define HAVE_ASYNC_LDS 1
#endif
#if __has_builtin(__builtin_amdgcn_s_wait_asynccnt)
#define HAVE_WAIT_ASYNC 1
#endif
#endif
#ifndef HAVE_ASYNC_LDS
#define HAVE_ASYNC_LDS 0
#endif
#ifndef HAVE_WAIT_ASYNC
#define HAVE_WAIT_ASYNC 0
#endif

// ---------------- vector / bf16 helpers ----------------
typedef __attribute__((ext_vector_type(16))) __bf16 bf16x16;
typedef __attribute__((ext_vector_type(8)))  __bf16 bf16x8;
typedef __attribute__((ext_vector_type(8)))  float  f32x8;

__device__ __forceinline__ unsigned short f2bf(float f) {
  unsigned u = __builtin_bit_cast(unsigned, f);
  u += 0x7FFFu + ((u >> 16) & 1u);               // round-to-nearest-even
  return (unsigned short)(u >> 16);
}

__device__ __forceinline__ bf16x16 load_frag_pair(const unsigned short* p0,
                                                  const unsigned short* p1) {
  bf16x8 lo = *reinterpret_cast<const bf16x8*>(p0);
  bf16x8 hi = *reinterpret_cast<const bf16x8*>(p1);
  bf16x16 v;
#pragma unroll
  for (int i = 0; i < 8; ++i) { v[i] = lo[i]; v[i + 8] = hi[i]; }
  return v;
}

// ---------------- kernel 1: W1,W2 -> bf16 transposed [N][K] ----------------
__global__ __launch_bounds__(256) void convert_weights_kernel(
    const float* __restrict__ W1, const float* __restrict__ W2,
    unsigned short* __restrict__ w1t, unsigned short* __restrict__ w2t) {
  const int i = blockIdx.x * 256 + threadIdx.x;           // 786432 total
  if (i < NFEAT_ * HID_) {                                // W1T: [HID][NFEAT]
    const int n = i / NFEAT_, k = i % NFEAT_;
    w1t[i] = f2bf(W1[(size_t)k * HID_ + n]);
  } else {                                                // W2T: [NCLASS][HID]
    const int j = i - NFEAT_ * HID_;
    const int n = j / HID_, k = j % HID_;
    w2t[j] = f2bf(W2[(size_t)k * NCLASS_ + n]);
  }
}

// ---------------- kernel 2: fused MLP, 32 rows per workgroup ----------------
// h = relu(x@W1+b1) kept in LDS as bf16; z_node = h@W2+b2 written f32.
// 2 M-tiles per wave iteration: each B fragment feeds two WMMAs (halves L2
// traffic on W1T/W2T, the dominant non-HBM cost).
__global__ __launch_bounds__(256) void mlp_fused_kernel(
    const float* __restrict__ x,
    const unsigned short* __restrict__ w1t, const float* __restrict__ b1,
    const unsigned short* __restrict__ w2t, const float* __restrict__ b2,
    float* __restrict__ z_node) {
  // region A (64KB): f32 x staging, later reused as bf16 h tile
  // region B (32KB): bf16 x tile
  __shared__ __align__(16) unsigned char smem[MROWS_ * HID_ * 2 + MROWS_ * NFEAT_ * 2];
  float*          xsf = reinterpret_cast<float*>(smem);                  // 32x512 f32
  unsigned short* hs  = reinterpret_cast<unsigned short*>(smem);         // 32x1024 bf16
  unsigned short* xs  = reinterpret_cast<unsigned short*>(smem + MROWS_ * HID_ * 2);

  const int m0  = blockIdx.x * MROWS_;
  const int tid = threadIdx.x;
  const float* xrow = x + (size_t)m0 * NFEAT_;   // contiguous 16384 floats

#if HAVE_ASYNC_LDS
  // stage f32 tile via async DMA (ASYNCcnt), then convert LDS->LDS to bf16
  typedef int i32x4_t __attribute__((__vector_size__(16)));
  typedef __attribute__((address_space(1))) i32x4_t* gvec_p;
  typedef __attribute__((address_space(3))) i32x4_t* lvec_p;
  for (int i4 = tid; i4 < (MROWS_ * NFEAT_) / 4; i4 += 256) {
    __builtin_amdgcn_global_load_async_to_lds_b128(
        (gvec_p)(xrow + (size_t)i4 * 4),
        (lvec_p)(xsf + (size_t)i4 * 4), 0, 0);
  }
#if HAVE_WAIT_ASYNC
  __builtin_amdgcn_s_wait_asynccnt(0);
#else
  asm volatile("s_wait_asynccnt 0x0" ::: "memory");
#endif
  __syncthreads();
  for (int i = tid; i < MROWS_ * NFEAT_; i += 256) xs[i] = f2bf(xsf[i]);
#else
  (void)xsf;
  for (int i = tid; i < MROWS_ * NFEAT_; i += 256) xs[i] = f2bf(xrow[i]);
#endif
  __syncthreads();

  const int wave  = tid >> 5;
  const int lane  = tid & 31;
  const int lhalf = lane >> 4;          // 0 | 1
  const int l16   = lane & 15;
  const int koffA = lhalf ? 8 : 0;      // A-frag K split per ISA layout
  const int koffB = lhalf ? 16 : 0;     // unused; B frags use koffA, same k order as A

  // ---- GEMM1: each wave owns 8 N-tiles of HID, 2 M-tiles each ----
  for (int t = 0; t < 8; ++t) {
    const int n = (wave * 8 + t) * 16 + l16;
    f32x8 acc0 = {0.f, 0.f, 0.f, 0.f, 0.f, 0.f, 0.f, 0.f};
    f32x8 acc1 = {0.f, 0.f, 0.f, 0.f, 0.f, 0.f, 0.f, 0.f};
    const unsigned short* arow0 = xs + l16 * NFEAT_;
    const unsigned short* arow1 = xs + (16 + l16) * NFEAT_;
    const unsigned short* brow  = w1t + (size_t)n * NFEAT_;
#pragma unroll 2
    for (int kk = 0; kk < NFEAT_ / 32; ++kk) {
      const int k0 = kk * 32;
      bf16x16 b  = load_frag_pair(brow + k0 + koffA, brow + k0 + 16 + koffA);
      bf16x16 a0 = load_frag_pair(arow0 + k0 + koffA, arow0 + k0 + 16 + koffA);
      bf16x16 a1 = load_frag_pair(arow1 + k0 + koffA, arow1 + k0 + 16 + koffA);
      acc0 = __builtin_amdgcn_wmma_f32_16x16x32_bf16(false, a0, false, b,
                                                     (short)0, acc0, false, false);
      acc1 = __builtin_amdgcn_wmma_f32_16x16x32_bf16(false, a1, false, b,
                                                     (short)0, acc1, false, false);
    }
    const float bias = b1[n];
#pragma unroll
    for (int v = 0; v < 8; ++v) {
      const int row = v + (lhalf << 3);
      float h0 = acc0[v] + bias;
      float h1 = acc1[v] + bias;
      hs[row * HID_ + n]        = f2bf(h0 > 0.f ? h0 : 0.f);
      hs[(16 + row) * HID_ + n] = f2bf(h1 > 0.f ? h1 : 0.f);
    }
  }
  __syncthreads();

  // ---- GEMM2: each wave owns 2 N-tiles of NCLASS, 2 M-tiles each ----
  for (int t = 0; t < 2; ++t) {
    const int n = (wave * 2 + t) * 16 + l16;
    f32x8 acc0 = {0.f, 0.f, 0.f, 0.f, 0.f, 0.f, 0.f, 0.f};
    f32x8 acc1 = {0.f, 0.f, 0.f, 0.f, 0.f, 0.f, 0.f, 0.f};
    const unsigned short* arow0 = hs + l16 * HID_;
    const unsigned short* arow1 = hs + (16 + l16) * HID_;
    const unsigned short* brow  = w2t + (size_t)n * HID_;
#pragma unroll 2
    for (int kk = 0; kk < HID_ / 32; ++kk) {
      const int k0 = kk * 32;
      bf16x16 b  = load_frag_pair(brow + k0 + koffA, brow + k0 + 16 + koffA);
      bf16x16 a0 = load_frag_pair(arow0 + k0 + koffA, arow0 + k0 + 16 + koffA);
      bf16x16 a1 = load_frag_pair(arow1 + k0 + koffA, arow1 + k0 + 16 + koffA);
      acc0 = __builtin_amdgcn_wmma_f32_16x16x32_bf16(false, a0, false, b,
                                                     (short)0, acc0, false, false);
      acc1 = __builtin_amdgcn_wmma_f32_16x16x32_bf16(false, a1, false, b,
                                                     (short)0, acc1, false, false);
    }
    const float bias = b2[n];
#pragma unroll
    for (int v = 0; v < 8; ++v) {
      const int row = v + (lhalf << 3);
      z_node[(size_t)(m0 + row) * NCLASS_ + n]      = acc0[v] + bias;
      z_node[(size_t)(m0 + 16 + row) * NCLASS_ + n] = acc1[v] + bias;
    }
  }
}

// ---------------- kernel 3: mean-pool per subgraph + sigmoid score ----------
__global__ __launch_bounds__(256) void pool_score_kernel(
    const float* __restrict__ z_node, const float* __restrict__ proj,
    float* __restrict__ z_sub, float* __restrict__ score) {
  __shared__ float r1[256], r2[256];
  const int s = blockIdx.x, c = threadIdx.x;
  float sum = 0.f;
  const float* base = z_node + (size_t)s * NPER_ * NCLASS_ + c;
#pragma unroll 4
  for (int n = 0; n < NPER_; ++n) sum += base[(size_t)n * NCLASS_];
  sum *= (1.f / (float)NPER_);
  z_sub[(size_t)s * NCLASS_ + c] = sum;
  const float p = proj[c];
  r1[c] = sum * p;
  r2[c] = p * p;
  __syncthreads();
  for (int off = 128; off > 0; off >>= 1) {
    if (c < off) { r1[c] += r1[c + off]; r2[c] += r2[c + off]; }
    __syncthreads();
  }
  if (c == 0) {
    const float t = r1[0] / sqrtf(r2[0]);
    score[s] = 1.f / (1.f + expf(-t));
  }
}

// ---------------- kernel 4: per-graph top-K via rank + compaction ----------
__global__ __launch_bounds__(64) void topk_kernel(
    const float* __restrict__ score, const int* __restrict__ belong,
    int* __restrict__ sub_index, int* __restrict__ sub_mask,
    float* __restrict__ out_score, float* __restrict__ out_belong) {
  __shared__ float sh[S_];
  __shared__ int   keep[S_];
  const int g = blockIdx.x, s = threadIdx.x;
  const float sc = score[g * S_ + s];
  sh[s] = sc;
  __syncthreads();
  int rank = 0;
  for (int j = 0; j < S_; ++j) {
    const float o = sh[j];
    rank += (o > sc) || (o == sc && j < s);   // top_k tie-break: lower idx first
  }
  keep[s] = (rank < KTOP_) ? 1 : 0;
  __syncthreads();
  int pos = 0;
  for (int j = 0; j < s; ++j) pos += keep[j];
  const int gs = g * S_ + s;
  if (keep[s]) {
    const int gi = g * KTOP_ + pos;           // ascending s -> ascending gi
    sub_index[gi]  = gs;
    sub_mask[gs]   = gi;
    out_score[gi]  = sc;
    out_belong[gi] = (float)belong[gs];
  } else {
    sub_mask[gs] = -1;
  }
}

// ---------------- kernel 5: new_x gather (float4) ----------------
__global__ __launch_bounds__(256) void gather_x_kernel(
    const float* __restrict__ x, const int* __restrict__ sub_index,
    float* __restrict__ out) {
  const size_t i4 = (size_t)blockIdx.x * 256 + threadIdx.x;  // NEWN_*128
  const int row  = (int)(i4 >> 7);
  const int col4 = (int)(i4 & 127);
  const int sub  = sub_index[row >> 5];
  const size_t src = ((size_t)(sub * NPER_ + (row & 31)) * NFEAT_) + (size_t)col4 * 4;
  *reinterpret_cast<float4*>(out + i4 * 4) =
      *reinterpret_cast<const float4*>(x + src);
}

// ---------------- kernel 6: z_node gather (float4) ----------------
__global__ __launch_bounds__(256) void gather_z_kernel(
    const float* __restrict__ z_node, const int* __restrict__ sub_index,
    float* __restrict__ out) {
  const size_t i4 = (size_t)blockIdx.x * 256 + threadIdx.x;  // NEWN_*64
  const int row  = (int)(i4 >> 6);
  const int col4 = (int)(i4 & 63);
  const int sub  = sub_index[row >> 5];
  const size_t src = ((size_t)(sub * NPER_ + (row & 31)) * NCLASS_) + (size_t)col4 * 4;
  *reinterpret_cast<float4*>(out + i4 * 4) =
      *reinterpret_cast<const float4*>(z_node + src);
}

// ---------------- kernel 7: z_sub gather ----------------
__global__ __launch_bounds__(256) void gather_zsub_kernel(
    const float* __restrict__ z_sub, const int* __restrict__ sub_index,
    float* __restrict__ out) {
  const int j = blockIdx.x, c = threadIdx.x;
  out[(size_t)j * NCLASS_ + c] = z_sub[(size_t)sub_index[j] * NCLASS_ + c];
}

// ---------------- kernel 8: batch / mapping gather ----------------
__global__ __launch_bounds__(256) void gather_meta_kernel(
    const int* __restrict__ batch, const int* __restrict__ mapping,
    const int* __restrict__ sub_index, const int* __restrict__ sub_mask,
    float* __restrict__ out_batch, float* __restrict__ out_mapping) {
  const int i = blockIdx.x * 256 + threadIdx.x;              // NEWN_
  const int sub = sub_index[i >> 5];
  const int src = sub * NPER_ + (i & 31);
  out_batch[i]   = (float)sub_mask[batch[src]];
  out_mapping[i] = (float)mapping[src];
}

// ---------------- kernel 9: edge remap ----------------
__global__ __launch_bounds__(256) void edges_kernel(
    const int* __restrict__ ei, const int* __restrict__ sub_mask,
    float* __restrict__ out) {
  const int e = blockIdx.x * 256 + threadIdx.x;              // E_
  const int u = ei[e];
  const int v = ei[E_ + e];
  const int mu_s = sub_mask[u >> 5];
  const int mv_s = sub_mask[v >> 5];
  const int mu = (mu_s < 0) ? -1 : (mu_s * NPER_ + (u & 31));
  const int mv = (mv_s < 0) ? -1 : (mv_s * NPER_ + (v & 31));
  const bool valid = (mu >= 0) && (mv >= 0);
  out[e]      = valid ? (float)mu : -1.f;
  out[E_ + e] = valid ? (float)mv : -1.f;
}

// ---------------- launch ----------------
extern "C" void kernel_launch(void* const* d_in, const int* in_sizes, int n_in,
                              void* d_out, int out_size, void* d_ws, size_t ws_size,
                              hipStream_t stream) {
  (void)in_sizes; (void)n_in; (void)out_size; (void)ws_size;
  const float* x       = (const float*)d_in[0];
  const int*   ei      = (const int*)d_in[1];
  const int*   batch   = (const int*)d_in[2];
  const int*   belong  = (const int*)d_in[3];
  const int*   mapping = (const int*)d_in[4];
  const float* W1      = (const float*)d_in[5];
  const float* b1      = (const float*)d_in[6];
  const float* W2      = (const float*)d_in[7];
  const float* b2      = (const float*)d_in[8];
  const float* proj    = (const float*)d_in[9];
  float* out = (float*)d_out;

  // workspace layout (256B aligned)
  char* ws = (char*)d_ws;
  constexpr size_t SZ_W1T  = (size_t)NFEAT_ * HID_ * 2;        // 1 MB
  constexpr size_t SZ_W2T  = (size_t)HID_ * NCLASS_ * 2;       // 0.5 MB
  constexpr size_t SZ_ZN   = (size_t)NNODES_ * NCLASS_ * 4;    // 134 MB
  constexpr size_t SZ_ZS   = (size_t)NSUBS_ * NCLASS_ * 4;     // 4 MB
  constexpr size_t SZ_SC   = (size_t)NSUBS_ * 4;
  constexpr size_t SZ_SI   = (size_t)NEWSUBS_ * 4;
  size_t off = 0;
  auto take = [&](size_t b) { size_t o = off; off = (off + b + 255) & ~(size_t)255; return o; };
  unsigned short* w1t   = (unsigned short*)(ws + take(SZ_W1T));
  unsigned short* w2t   = (unsigned short*)(ws + take(SZ_W2T));
  float* z_node         = (float*)(ws + take(SZ_ZN));
  float* z_sub          = (float*)(ws + take(SZ_ZS));
  float* score          = (float*)(ws + take(SZ_SC));
  int*   sub_index      = (int*)  (ws + take(SZ_SI));
  int*   sub_mask       = (int*)  (ws + take((size_t)NSUBS_ * 4));

  convert_weights_kernel<<<(NFEAT_ * HID_ + HID_ * NCLASS_) / 256, 256, 0, stream>>>(
      W1, W2, w1t, w2t);
  mlp_fused_kernel<<<NNODES_ / MROWS_, 256, 0, stream>>>(x, w1t, b1, w2t, b2, z_node);
  pool_score_kernel<<<NSUBS_, 256, 0, stream>>>(z_node, proj, z_sub, score);
  topk_kernel<<<G_, 64, 0, stream>>>(score, belong, sub_index, sub_mask,
                                     out + O_SCORE, out + O_BELON);
  gather_x_kernel<<<(NEWN_ * (NFEAT_ / 4)) / 256, 256, 0, stream>>>(
      x, sub_index, out + O_NEWX);
  gather_z_kernel<<<(NEWN_ * (NCLASS_ / 4)) / 256, 256, 0, stream>>>(
      z_node, sub_index, out + O_ZNODE);
  gather_zsub_kernel<<<NEWSUBS_, 256, 0, stream>>>(z_sub, sub_index, out + O_ZSUB);
  gather_meta_kernel<<<NEWN_ / 256, 256, 0, stream>>>(
      batch, mapping, sub_index, sub_mask, out + O_BATCH, out + O_MAP);
  edges_kernel<<<E_ / 256, 256, 0, stream>>>(ei, sub_mask, out + O_EDGE);
}
